// OwnLSTM_75290776699345
// MI455X (gfx1250) — matrix operand order repaired and checked by reading, not compiled, not run on hardware
//
#include <hip/hip_runtime.h>

typedef __attribute__((ext_vector_type(16))) __bf16 v16bf;
typedef __attribute__((ext_vector_type(8)))  __bf16 v8bf;
typedef __attribute__((ext_vector_type(8)))  float  v8f;

#define NB   128            // batch
#define NT   256            // time steps
#define NIN  512            // input size
#define NH   1024           // hidden size
#define NKT  1536           // NIN + NH
#define NOUT 512            // output size

// ---------- helpers ----------

__device__ __forceinline__ __bf16 f2bf(float f) {
    unsigned u = __builtin_bit_cast(unsigned, f);
    unsigned r = u + 0x7FFFu + ((u >> 16) & 1u);   // round-to-nearest-even
    unsigned short s = (unsigned short)(r >> 16);
    return __builtin_bit_cast(__bf16, s);
}

__device__ __forceinline__ float fsigmoid(float x) {
    return 1.0f / (1.0f + __expf(-x));
}

__device__ __forceinline__ float ftanh(float x) {
    float e = __expf(2.0f * x);
    return (e - 1.0f) / (e + 1.0f);
}

// A fragment (16x32 bf16, M x K): lane l<16 holds row M=l, K = k0..k0+15;
// lanes 16..31 hold same rows with K = k0+16..k0+31 (contiguous per lane).
// caller passes p = row_ptr + k0 + half*16 ; chunks at p and p+8.
__device__ __forceinline__ v16bf load_frag_a(const __bf16* p) {
    v8bf lo = *reinterpret_cast<const v8bf*>(p);
    v8bf hi = *reinterpret_cast<const v8bf*>(p + 8);
    return __builtin_shufflevector(lo, hi, 0,1,2,3,4,5,6,7,8,9,10,11,12,13,14,15);
}

// B fragment (32x16 bf16, K x N) from row-major W[N][K]: lane l holds column N=l,
// lanes 0..15 K=k0..k0+15, lanes 16..31 K=k0+16..k0+31 (contiguous per lane).
// caller passes p = Wrow_ptr + k0 + half*16 ; chunks at p and p+8.
__device__ __forceinline__ v16bf load_frag_b(const __bf16* p) {
    v8bf lo = *reinterpret_cast<const v8bf*>(p);
    v8bf hi = *reinterpret_cast<const v8bf*>(p + 8);
    return __builtin_shufflevector(lo, hi, 0,1,2,3,4,5,6,7,8,9,10,11,12,13,14,15);
}

__device__ __forceinline__ v8f wmma_bf16(v16bf a, v16bf b, v8f c) {
    return __builtin_amdgcn_wmma_f32_16x16x32_bf16(
        /*neg_a=*/false, a, /*neg_b=*/false, b,
        /*c_mod=*/(short)0, c, /*reuse_a=*/false, /*reuse_b=*/false);
}

// ---------- utility kernels ----------

__global__ void cvt_f32_bf16(const float* __restrict__ src, __bf16* __restrict__ dst, int n) {
    int i = blockIdx.x * blockDim.x + threadIdx.x;
    int stride = gridDim.x * blockDim.x;
    for (; i < n; i += stride) dst[i] = f2bf(src[i]);
}

__global__ void zero_bytes16(unsigned* __restrict__ dst, int n_u4) {
    // dst treated as uint4-granular region, n_u4 = bytes/16
    int i = blockIdx.x * blockDim.x + threadIdx.x;
    int stride = gridDim.x * blockDim.x;
    uint4* p = reinterpret_cast<uint4*>(dst);
    uint4 z; z.x = z.y = z.z = z.w = 0u;
    for (; i < n_u4; i += stride) p[i] = z;
}

// ---------- fused LSTM step: G = [x_t|h] @ Wg^T + b ; cell update ----------
// Wg is [4*NH][NKT] bf16, gate order f,i,c,o.  Each wave owns one 16x16 tile
// of (batch x hidden-unit), for ALL 4 gates, so the cell update is fused.
__global__ __launch_bounds__(256) void lstm_step(
    const __bf16* __restrict__ xbf,   // [NB][NT][NIN]
    const __bf16* __restrict__ Wg,    // [4*NH][NKT]
    const float* __restrict__ bf_g, const float* __restrict__ bi_g,
    const float* __restrict__ bc_g, const float* __restrict__ bo_g,
    const __bf16* __restrict__ hprev, // [NB][NH]
    __bf16* __restrict__ hnext,       // [NB][NH]
    float*  __restrict__ cstate,      // [NB][NH]
    __bf16* __restrict__ hs,          // [NB][NT][NH]
    int t)
{
    const int wid  = blockIdx.x * (blockDim.x >> 5) + (threadIdx.x >> 5);
    const int lane = threadIdx.x & 31;
    const int jt = wid & (NH / 16 - 1);   // 0..63 : hidden tile
    const int mt = wid >> 6;              // 0..7  : batch tile
    const int half = lane >> 4;
    const int l = lane & 15;
    const int m = mt * 16 + l;            // batch row this lane loads (A)
    const int j0 = jt * 16;
    const int n = j0 + l;                 // hidden unit this lane loads (B)

    // bias-initialized accumulators
    v8f accF, accI, accC, accO;
    {
        float vf = bf_g[n], vi = bi_g[n], vc = bc_g[n], vo = bo_g[n];
        #pragma unroll
        for (int r = 0; r < 8; r++) { accF[r] = vf; accI[r] = vi; accC[r] = vc; accO[r] = vo; }
    }

    const __bf16* arow_x = xbf + ((size_t)m * NT + t) * NIN + half * 16;
    const __bf16* arow_h = hprev + (size_t)m * NH + half * 16;
    const __bf16* browF = Wg + (size_t)(0 * NH + n) * NKT + half * 16;
    const __bf16* browI = Wg + (size_t)(1 * NH + n) * NKT + half * 16;
    const __bf16* browC = Wg + (size_t)(2 * NH + n) * NKT + half * 16;
    const __bf16* browO = Wg + (size_t)(3 * NH + n) * NKT + half * 16;

    // K over x_t: columns 0..NIN-1 of Wg
    #pragma unroll 2
    for (int k0 = 0; k0 < NIN; k0 += 32) {
        v16bf a  = load_frag_a(arow_x + k0);
        v16bf bF = load_frag_b(browF + k0);
        v16bf bI = load_frag_b(browI + k0);
        v16bf bC = load_frag_b(browC + k0);
        v16bf bO = load_frag_b(browO + k0);
        accF = wmma_bf16(a, bF, accF);
        accI = wmma_bf16(a, bI, accI);
        accC = wmma_bf16(a, bC, accC);
        accO = wmma_bf16(a, bO, accO);
    }
    // K over h_{t-1}: columns NIN..NKT-1 of Wg
    #pragma unroll 2
    for (int k0 = 0; k0 < NH; k0 += 32) {
        v16bf a  = load_frag_a(arow_h + k0);
        v16bf bF = load_frag_b(browF + NIN + k0);
        v16bf bI = load_frag_b(browI + NIN + k0);
        v16bf bC = load_frag_b(browC + NIN + k0);
        v16bf bO = load_frag_b(browO + NIN + k0);
        accF = wmma_bf16(a, bF, accF);
        accI = wmma_bf16(a, bI, accI);
        accC = wmma_bf16(a, bC, accC);
        accO = wmma_bf16(a, bO, accO);
    }

    // fused cell update. C/D layout: VGPR r, lanes 0-15 -> M=r, lanes 16-31 -> M=r+8, N = lane&15.
    #pragma unroll
    for (int r = 0; r < 8; r++) {
        int mr = mt * 16 + r + half * 8;
        size_t idx = (size_t)mr * NH + n;
        float fg = fsigmoid(accF[r]);
        float ig = fsigmoid(accI[r]);
        float ct = ftanh(accC[r]);
        float og = fsigmoid(accO[r]);
        float cn = fg * cstate[idx] + ig * ct;
        cstate[idx] = cn;
        float hv = og * ftanh(cn);
        __bf16 hb = f2bf(hv);
        hnext[idx] = hb;
        hs[((size_t)mr * NT + t) * NH + n] = hb;
    }
}

// ---------- output projection: out = hs @ W_out^T + b_out ----------
// Each wave computes a 16(M) x 64(N) block (4 independent WMMA chains).
__global__ __launch_bounds__(256) void out_proj(
    const __bf16* __restrict__ hs,    // [NB*NT][NH]
    const __bf16* __restrict__ Wo,    // [NOUT][NH]
    const float* __restrict__ bout,   // [NOUT]
    float* __restrict__ out)          // [NB*NT][NOUT]
{
    const int wid  = blockIdx.x * (blockDim.x >> 5) + (threadIdx.x >> 5);
    const int lane = threadIdx.x & 31;
    const int half = lane >> 4;
    const int l = lane & 15;
    const int ng = wid & 7;            // N-group: 64 columns
    const int mt = wid >> 3;           // 0..2047
    const int m  = mt * 16 + l;
    const int nbase = ng * 64;

    const __bf16* arow = hs + (size_t)m * NH + half * 16;
    const __bf16* brow[4];
    v8f acc[4];
    #pragma unroll
    for (int q = 0; q < 4; q++) {
        int nq = nbase + q * 16 + l;
        brow[q] = Wo + (size_t)nq * NH + half * 16;
        float b = bout[nq];
        #pragma unroll
        for (int r = 0; r < 8; r++) acc[q][r] = b;
    }

    #pragma unroll 2
    for (int k0 = 0; k0 < NH; k0 += 32) {
        v16bf a = load_frag_a(arow + k0);
        #pragma unroll
        for (int q = 0; q < 4; q++) {
            v16bf b = load_frag_b(brow[q] + k0);
            acc[q] = wmma_bf16(a, b, acc[q]);
        }
    }

    #pragma unroll
    for (int q = 0; q < 4; q++) {
        #pragma unroll
        for (int r = 0; r < 8; r++) {
            int mr = mt * 16 + r + half * 8;
            int nq = nbase + q * 16 + l;
            out[(size_t)mr * NOUT + nq] = acc[q][r];
        }
    }
}

// ---------- host side ----------

static inline size_t align256(size_t x) { return (x + 255) & ~(size_t)255; }

extern "C" void kernel_launch(void* const* d_in, const int* in_sizes, int n_in,
                              void* d_out, int out_size, void* d_ws, size_t ws_size,
                              hipStream_t stream) {
    const float* x     = (const float*)d_in[0];
    const float* W_f   = (const float*)d_in[1];
    const float* b_f   = (const float*)d_in[2];
    const float* W_i   = (const float*)d_in[3];
    const float* b_i   = (const float*)d_in[4];
    const float* W_c   = (const float*)d_in[5];
    const float* b_c   = (const float*)d_in[6];
    const float* W_o   = (const float*)d_in[7];
    const float* b_o   = (const float*)d_in[8];
    const float* W_out = (const float*)d_in[9];
    const float* b_out = (const float*)d_in[10];
    float* out = (float*)d_out;

    // workspace layout
    char* ws = (char*)d_ws;
    size_t off = 0;
    __bf16* xbf  = (__bf16*)(ws + off); off = align256(off + (size_t)NB * NT * NIN * 2);
    __bf16* Wg   = (__bf16*)(ws + off); off = align256(off + (size_t)4 * NH * NKT * 2);
    __bf16* Wobf = (__bf16*)(ws + off); off = align256(off + (size_t)NOUT * NH * 2);
    __bf16* hA   = (__bf16*)(ws + off); size_t zero_begin = off;
                                        off = align256(off + (size_t)NB * NH * 2);
    __bf16* hB   = (__bf16*)(ws + off); off = align256(off + (size_t)NB * NH * 2);
    float*  cst  = (float*) (ws + off); off = align256(off + (size_t)NB * NH * 4);
    size_t zero_end = off;
    __bf16* hs   = (__bf16*)(ws + off); off = align256(off + (size_t)NB * NT * NH * 2);
    (void)ws_size;

    // one-time conversions to bf16 (gate order f,i,c,o fused row-wise)
    {
        int n = NB * NT * NIN;
        cvt_f32_bf16<<<2048, 256, 0, stream>>>(x, xbf, n);
        int nw = NH * NKT;
        cvt_f32_bf16<<<1024, 256, 0, stream>>>(W_f, Wg + (size_t)0 * NH * NKT, nw);
        cvt_f32_bf16<<<1024, 256, 0, stream>>>(W_i, Wg + (size_t)1 * NH * NKT, nw);
        cvt_f32_bf16<<<1024, 256, 0, stream>>>(W_c, Wg + (size_t)2 * NH * NKT, nw);
        cvt_f32_bf16<<<1024, 256, 0, stream>>>(W_o, Wg + (size_t)3 * NH * NKT, nw);
        cvt_f32_bf16<<<512, 256, 0, stream>>>(W_out, Wobf, NOUT * NH);
        // zero h0 (both ping-pong halves) and c0
        int n_u4 = (int)((zero_end - zero_begin) / 16);
        zero_bytes16<<<256, 256, 0, stream>>>((unsigned*)(ws + zero_begin), n_u4);
    }

    // sequential recurrence: 256 step kernels, h ping-pongs
    for (int t = 0; t < NT; t++) {
        const __bf16* hp = (t & 1) ? hB : hA;
        __bf16*       hn = (t & 1) ? hA : hB;
        // 512 waves = 8 batch-tiles x 64 hidden-tiles; 8 waves/block
        lstm_step<<<64, 256, 0, stream>>>(xbf, Wg, b_f, b_i, b_c, b_o,
                                          hp, hn, cst, hs, t);
    }

    // final projection: 16384 waves (2048 blocks x 8 waves)
    out_proj<<<2048, 256, 0, stream>>>(hs, Wobf, b_out, out);
}
